// LatentPredictor_12206297055447
// MI455X (gfx1250) — matrix operand order repaired and checked by reading, not compiled, not run on hardware
//
#include <hip/hip_runtime.h>

// ---------------------------------------------------------------------------
// Problem constants (match reference)
// ---------------------------------------------------------------------------
#define B_   64
#define T_   302
#define D_   768
#define H_   8
#define DH_  96
#define L_   4
#define W_   10
#define M_   (B_ * T_)        // 19328 rows = 151 * 128

typedef __bf16 bf16;
typedef bf16  v16bf __attribute__((ext_vector_type(16)));
typedef float v8f   __attribute__((ext_vector_type(8)));
typedef unsigned int u32x4 __attribute__((ext_vector_type(4)));
typedef int i32x4 __attribute__((ext_vector_type(4)));
typedef int i32x8 __attribute__((ext_vector_type(8)));

// GEMM tiling: 256 threads = 8 wave32; waves arranged 4 (M) x 2 (N),
// each wave computes a 32x64 tile as 2x4 WMMA 16x16 accumulators.
#define BM 128
#define BN 128
#define BK 32
#define LDSK 40   // 32 bf16 row (64B) + 16B pad -> matches TDM pad(16dw,4dw)

// Tensor Data Mover path (CDNA5): HBM/L2 -> LDS DMA tracked by TENSORcnt.
#if defined(__AMDGCN__) && __has_builtin(__builtin_amdgcn_tensor_load_to_lds) && \
    __has_builtin(__builtin_amdgcn_s_wait_tensorcnt)
#define USE_TDM 1
#else
#define USE_TDM 0
#endif

// ---------------------------------------------------------------------------
// Weight prep: fp32 W[l][k][n] -> bf16 Wt[(l*4+which)][n][k]  (LDS-tiled)
// ---------------------------------------------------------------------------
__global__ __launch_bounds__(256)
void transpose_cvt(const float* __restrict__ Wsrc, bf16* __restrict__ WtBase,
                   int which) {
  __shared__ float tile[32][33];
  const int tilesPerMat = (D_ / 32) * (D_ / 32);
  const int l  = blockIdx.x / tilesPerMat;
  const int tl = blockIdx.x % tilesPerMat;
  const int tk = tl % (D_ / 32);
  const int tn = tl / (D_ / 32);
  const float* W = Wsrc + (size_t)l * D_ * D_;
  bf16* Wt = WtBase + ((size_t)l * 4 + which) * D_ * D_;

  const int tx = threadIdx.x & 31;
  const int ty = threadIdx.x >> 5;   // 0..7
#pragma unroll
  for (int r = 0; r < 32; r += 8) {
    const int k = tk * 32 + ty + r;
    const int n = tn * 32 + tx;
    tile[ty + r][tx] = W[(size_t)k * D_ + n];
  }
  __syncthreads();
#pragma unroll
  for (int r = 0; r < 32; r += 8) {
    const int n = tn * 32 + ty + r;
    const int k = tk * 32 + tx;
    Wt[(size_t)n * D_ + k] = (bf16)tile[tx][ty + r];
  }
}

// ---------------------------------------------------------------------------
// h = bf16(x + pos)
// ---------------------------------------------------------------------------
__global__ __launch_bounds__(256)
void addpos_cvt(const float* __restrict__ x, const float* __restrict__ pos,
                bf16* __restrict__ hb) {
  const size_t i = (size_t)blockIdx.x * blockDim.x + threadIdx.x;
  const int d = (int)(i % D_);
  const int t = (int)((i / D_) % T_);
  hb[i] = (bf16)(x[i] + pos[(size_t)t * D_ + d]);
}

#if USE_TDM
// Issue one TDM 2D tile load: 32 (K, contiguous bf16) x 128 rows, row stride
// 768 elements, into LDS at lds_off with 16B pad after each 64B row (LDSK=40).
// D# group0: count=1 | lds_addr | global_addr(57b) | type=2.
// D# group1: data_size=1(2B), pad_enable, pad_interval=3(16 dw), pad_amount=3(4 dw),
//            tensor_dim0=768, tensor_dim1=M_, tile_dim0=32, tile_dim1=128,
//            tensor_dim0_stride=768.
__device__ __forceinline__ void tdm_load_tile(unsigned int lds_off,
                                              const bf16* g) {
  const unsigned long long ga = (unsigned long long)(uintptr_t)g;
  u32x4 g0;
  g0[0] = 1u;                                   // count=1 (valid descriptor)
  g0[1] = lds_off;                              // LDS byte address
  g0[2] = (unsigned int)ga;                     // global_addr[31:0]
  g0[3] = ((unsigned int)(ga >> 32) & 0x01FFFFFFu) | 0x80000000u; // addr hi | type=2
  i32x8 g1;
  g1[0] = 0x06D10000;                           // 2B elems, pad 4dw every 16dw
  g1[1] = (int)(768u << 16);                    // tensor_dim0 = 768
  g1[2] = (int)(((unsigned)M_ & 0xFFFFu) << 16);// tensor_dim1 lo (OOB bound)
  g1[3] = (int)(32u << 16);                     // tile_dim0 = 32
  g1[4] = 128;                                  // tile_dim1 = 128 rows
  g1[5] = 768;                                  // tensor_dim0_stride
  g1[6] = 0;
  g1[7] = 0;
  i32x4 z4 = {0, 0, 0, 0};
#if defined(__clang_major__) && __clang_major__ >= 23
  i32x8 z8 = {0, 0, 0, 0, 0, 0, 0, 0};
  __builtin_amdgcn_tensor_load_to_lds(g0, g1, z4, z4, z8, 0);
#else
  __builtin_amdgcn_tensor_load_to_lds(g0, g1, z4, z4, 0);
#endif
}
#endif  // USE_TDM

// ---------------------------------------------------------------------------
// GEMM: C[M,N] = A[M,K](bf16) * Bt[N,K](bf16)^T + bias, via WMMA bf16.
// TDM feeds LDS (double buffered); all 8 waves spend their slots on WMMA.
// ---------------------------------------------------------------------------
template <bool OUT_BF16>
__global__ __launch_bounds__(256)
void gemm_bf16_wmma(const bf16* __restrict__ A, const bf16* __restrict__ Bt,
                    const float* __restrict__ bias,
                    float* __restrict__ C, bf16* __restrict__ Cb) {
  __shared__ bf16 As[2][BM][LDSK];
  __shared__ bf16 Bs[2][BN][LDSK];

  const int K = D_, N = D_;
  const int tid  = threadIdx.x;
  const int lane = tid & 31;
  const int wave = tid >> 5;
  const int wm   = wave >> 1;     // 0..3 -> 32-row strip
  const int wn   = wave & 1;      // 0..1 -> 64-col strip
  const int half = lane >> 4;
  const int l16  = lane & 15;

  const int tilesN  = N / BN;
  const int rowBase = (blockIdx.x / tilesN) * BM;
  const int colBase = (blockIdx.x % tilesN) * BN;

  v8f zero = {};
  v8f acc[2][4];
#pragma unroll
  for (int i = 0; i < 2; ++i)
#pragma unroll
    for (int j = 0; j < 4; ++j) acc[i][j] = zero;

  union Frag { uint4 u[2]; v16bf v; };
  const int nK = K / BK;

#if USE_TDM
  const bool issuer = (wave == 0);
  const bf16* Arow = A  + (size_t)rowBase * K;
  const bf16* Brow = Bt + (size_t)colBase * K;
  const unsigned int ldsA0 = (unsigned int)(uintptr_t)&As[0][0][0];
  const unsigned int ldsA1 = (unsigned int)(uintptr_t)&As[1][0][0];
  const unsigned int ldsB0 = (unsigned int)(uintptr_t)&Bs[0][0][0];
  const unsigned int ldsB1 = (unsigned int)(uintptr_t)&Bs[1][0][0];

  if (issuer) {
    tdm_load_tile(ldsA0, Arow);
    tdm_load_tile(ldsB0, Brow);
    __builtin_amdgcn_s_wait_tensorcnt(0);
  }
  __syncthreads();
#else
  // Fallback: per-thread register staging. A/B tiles are both 128x32.
  const int grow = tid >> 1;
  const int gkk  = (tid & 1) << 4;
  const bf16* Ag = A  + (size_t)(rowBase + grow) * K + gkk;
  const bf16* Bg = Bt + (size_t)(colBase + grow) * K + gkk;
  uint4 a0, a1, b0, b1;
  auto ldg = [&](int koff) {
    a0 = *reinterpret_cast<const uint4*>(Ag + koff);
    a1 = *reinterpret_cast<const uint4*>(Ag + koff + 8);
    b0 = *reinterpret_cast<const uint4*>(Bg + koff);
    b1 = *reinterpret_cast<const uint4*>(Bg + koff + 8);
  };
  auto sts = [&](int buf) {
    *reinterpret_cast<uint4*>(&As[buf][grow][gkk])     = a0;
    *reinterpret_cast<uint4*>(&As[buf][grow][gkk + 8]) = a1;
    *reinterpret_cast<uint4*>(&Bs[buf][grow][gkk])     = b0;
    *reinterpret_cast<uint4*>(&Bs[buf][grow][gkk + 8]) = b1;
  };
  ldg(0);
  sts(0);
  __syncthreads();
#endif

  for (int kt = 0; kt < nK; ++kt) {
    const int cur = kt & 1;
#if USE_TDM
    if (kt + 1 < nK && issuer) {
      const int ko = (kt + 1) * BK;
      tdm_load_tile(cur ? ldsA0 : ldsA1, Arow + ko);
      tdm_load_tile(cur ? ldsB0 : ldsB1, Brow + ko);
    }
#else
    if (kt + 1 < nK) ldg((kt + 1) * BK);
#endif

    // A fragment (ISA 16-bit A 16x32): lane half h holds K = h*8..h*8+7 and
    // 16+h*8..16+h*8+7 -> two 16B LDS reads.
    v16bf af[2], bfr[4];
#pragma unroll
    for (int i = 0; i < 2; ++i) {
      Frag f;
      const bf16* p = &As[cur][wm * 32 + i * 16 + l16][half * 8];
      f.u[0] = *reinterpret_cast<const uint4*>(p);
      f.u[1] = *reinterpret_cast<const uint4*>(p + 16);
      af[i] = f.v;
    }
    // B fragment: lane half h holds K = h*8..h*8+7 and 16+h*8..16+h*8+7.
#pragma unroll
    for (int j = 0; j < 4; ++j) {
      Frag f;
      const bf16* p = &Bs[cur][wn * 64 + j * 16 + l16][half * 8];
      f.u[0] = *reinterpret_cast<const uint4*>(p);
      f.u[1] = *reinterpret_cast<const uint4*>(p + 16);
      bfr[j] = f.v;
    }

#pragma unroll
    for (int i = 0; i < 2; ++i)
#pragma unroll
      for (int j = 0; j < 4; ++j)
        acc[i][j] = __builtin_amdgcn_wmma_f32_16x16x32_bf16(
            false, af[i], false, bfr[j], (short)0, acc[i][j], false, false);

#if USE_TDM
    if (kt + 1 < nK && issuer) __builtin_amdgcn_s_wait_tensorcnt(0);
#else
    if (kt + 1 < nK) sts(cur ^ 1);
#endif
    __syncthreads();
  }

  // Epilogue: C/D layout -> lane n = l16, VGPR r -> m = r + half*8. Fuse bias.
#pragma unroll
  for (int j = 0; j < 4; ++j) {
    const int n = colBase + wn * 64 + j * 16 + l16;
    const float bsv = bias[n];
#pragma unroll
    for (int i = 0; i < 2; ++i) {
      const int m0 = rowBase + wm * 32 + i * 16 + half * 8;
#pragma unroll
      for (int r = 0; r < 8; ++r) {
        const float val = acc[i][j][r] + bsv;
        if (OUT_BF16) Cb[(size_t)(m0 + r) * N + n] = (bf16)val;
        else          C [(size_t)(m0 + r) * N + n] = val;
      }
    }
  }
}

// ---------------------------------------------------------------------------
// Causal local-window attention: one wave32 per (b, t, h) query.
// Dh = 96 -> 3 floats per lane. q/k/v are bf16; fp32 math in registers.
// Output written as bf16 (feeds the out-projection GEMM directly).
// ---------------------------------------------------------------------------
__global__ __launch_bounds__(256)
void attn_local(const bf16* __restrict__ q, const bf16* __restrict__ kmat,
                const bf16* __restrict__ vmat, bf16* __restrict__ o) {
  const int lane = threadIdx.x & 31;
  const int gw   = blockIdx.x * 8 + (threadIdx.x >> 5);
  if (gw >= B_ * T_ * H_) return;
  const int hh = gw % H_;
  const int t  = (gw / H_) % T_;
  const int b  = gw / (H_ * T_);
  const size_t rowOff = ((size_t)b * T_ + t) * D_ + (size_t)hh * DH_;

  const float qr0 = (float)q[rowOff + lane];
  const float qr1 = (float)q[rowOff + lane + 32];
  const float qr2 = (float)q[rowOff + lane + 64];

  const int j0 = (t - W_ + 1 > 0) ? (t - W_ + 1) : 0;
  const int nj = t - j0 + 1;                    // 1..10 keys
  const float scale = 0.1020620726159658f;      // 1/sqrt(96)

  float sc[W_];
#pragma unroll
  for (int jj = 0; jj < W_; ++jj) {
    float s = -3.0e38f;
    if (jj < nj) {
      const size_t ko = ((size_t)b * T_ + (j0 + jj)) * D_ + (size_t)hh * DH_;
      float p = qr0 * (float)kmat[ko + lane]
              + qr1 * (float)kmat[ko + lane + 32]
              + qr2 * (float)kmat[ko + lane + 64];
#pragma unroll
      for (int off = 16; off > 0; off >>= 1) p += __shfl_xor(p, off, 32);
      s = p * scale;
    }
    sc[jj] = s;
  }

  float mx = sc[0];
#pragma unroll
  for (int jj = 1; jj < W_; ++jj) mx = fmaxf(mx, sc[jj]);
  float sum = 0.f;
#pragma unroll
  for (int jj = 0; jj < W_; ++jj) {
    const float e = (jj < nj) ? __expf(sc[jj] - mx) : 0.f;
    sc[jj] = e;
    sum += e;
  }
  const float inv = 1.f / sum;

  float a0 = 0.f, a1 = 0.f, a2 = 0.f;
  for (int jj = 0; jj < nj; ++jj) {
    const size_t vo = ((size_t)b * T_ + (j0 + jj)) * D_ + (size_t)hh * DH_;
    a0 += sc[jj] * (float)vmat[vo + lane];
    a1 += sc[jj] * (float)vmat[vo + lane + 32];
    a2 += sc[jj] * (float)vmat[vo + lane + 64];
  }
  o[rowOff + lane]      = (bf16)(a0 * inv);
  o[rowOff + lane + 32] = (bf16)(a1 * inv);
  o[rowOff + lane + 64] = (bf16)(a2 * inv);
}

// ---------------------------------------------------------------------------
// Host orchestration (graph-capture safe: only async launches on `stream`)
// ---------------------------------------------------------------------------
extern "C" void kernel_launch(void* const* d_in, const int* in_sizes, int n_in,
                              void* d_out, int out_size, void* d_ws, size_t ws_size,
                              hipStream_t stream) {
  (void)in_sizes; (void)n_in; (void)out_size; (void)ws_size;
  const float* x   = (const float*)d_in[0];
  const float* pos = (const float*)d_in[1];
  const float* Wq  = (const float*)d_in[2];
  const float* bq  = (const float*)d_in[3];
  const float* Wk  = (const float*)d_in[4];
  const float* bk  = (const float*)d_in[5];
  const float* Wv  = (const float*)d_in[6];
  const float* bv  = (const float*)d_in[7];
  const float* Wo  = (const float*)d_in[8];
  const float* bo  = (const float*)d_in[9];
  float* out = (float*)d_out;

  char* p = (char*)d_ws;
  auto carve = [&](size_t bytes) -> char* {
    char* r = p;
    p += (bytes + 255) & ~(size_t)255;
    return r;
  };
  bf16* Wt  = (bf16*)carve((size_t)16 * D_ * D_ * sizeof(bf16)); // 18.9 MB
  bf16* hbA = (bf16*)carve((size_t)M_ * D_ * sizeof(bf16));      // layer input
  bf16* hbB = (bf16*)carve((size_t)M_ * D_ * sizeof(bf16));      // attn output
  bf16* qb  = (bf16*)carve((size_t)M_ * D_ * sizeof(bf16));
  bf16* kb  = (bf16*)carve((size_t)M_ * D_ * sizeof(bf16));
  bf16* vb  = (bf16*)carve((size_t)M_ * D_ * sizeof(bf16));

  const int tgrid = L_ * (D_ / 32) * (D_ / 32);      // 2304
  transpose_cvt<<<tgrid, 256, 0, stream>>>(Wq, Wt, 0);
  transpose_cvt<<<tgrid, 256, 0, stream>>>(Wk, Wt, 1);
  transpose_cvt<<<tgrid, 256, 0, stream>>>(Wv, Wt, 2);
  transpose_cvt<<<tgrid, 256, 0, stream>>>(Wo, Wt, 3);

  addpos_cvt<<<(M_ * D_) / 256, 256, 0, stream>>>(x, pos, hbA);

  const int ggrid = (M_ / BM) * (D_ / BN);           // 151 * 6 = 906
  const int agrid = (B_ * T_ * H_) / 8;              // 19328

  const size_t DD = (size_t)D_ * D_;
  for (int l = 0; l < L_; ++l) {
    const bf16* wl = Wt + (size_t)l * 4 * DD;
    gemm_bf16_wmma<true><<<ggrid, 256, 0, stream>>>(hbA, wl + 0 * DD, bq + l * D_, nullptr, qb);
    gemm_bf16_wmma<true><<<ggrid, 256, 0, stream>>>(hbA, wl + 1 * DD, bk + l * D_, nullptr, kb);
    gemm_bf16_wmma<true><<<ggrid, 256, 0, stream>>>(hbA, wl + 2 * DD, bv + l * D_, nullptr, vb);
    attn_local<<<agrid, 256, 0, stream>>>(qb, kb, vb, hbB);
    if (l == L_ - 1)
      gemm_bf16_wmma<false><<<ggrid, 256, 0, stream>>>(hbB, wl + 3 * DD, bo + l * D_, out, nullptr);
    else
      gemm_bf16_wmma<true><<<ggrid, 256, 0, stream>>>(hbB, wl + 3 * DD, bo + l * D_, nullptr, hbA);
  }
}
